// ToxDL_GCN_Network_16381005267403
// MI455X (gfx1250) — matrix operand order repaired and checked by reading, not compiled, not run on hardware
//
#include <hip/hip_runtime.h>
#include <hip/hip_bf16.h>
#include <math.h>

// ---------------------------------------------------------------------------
// Problem constants (from reference)
// ---------------------------------------------------------------------------
#define N_NODES 10000
#define N_EDGES 160000
#define NGRAPH  64
#define D_IN    1280
#define D_HID   512
#define D_OUT   256

typedef __attribute__((ext_vector_type(16))) __bf16 v16bf;
typedef __attribute__((ext_vector_type(8)))  __bf16 v8bf;
typedef __attribute__((ext_vector_type(8)))  float  v8f;

// ---------------------------------------------------------------------------
// Conversions
// ---------------------------------------------------------------------------
__global__ void cvt_f32_to_bf16(const float* __restrict__ in,
                                __bf16* __restrict__ out, int n) {
    int i = blockIdx.x * blockDim.x + threadIdx.x;
    if (i < n) out[i] = (__bf16)in[i];
}

// Wt[n*K + k] = (bf16)W[k*N + n]   (store weights transposed so GEMM B-tile
// loads are contiguous along K)
__global__ void cvt_transpose_bf16(const float* __restrict__ W,
                                   __bf16* __restrict__ Wt, int K, int N) {
    int i = blockIdx.x * blockDim.x + threadIdx.x;
    if (i >= K * N) return;
    int k = i / N, n = i - k * N;
    Wt[(size_t)n * K + k] = (__bf16)W[i];
}

// ---------------------------------------------------------------------------
// Degree / normalization
// ---------------------------------------------------------------------------
__global__ void deg_init(float* __restrict__ deg, int n) {
    int i = blockIdx.x * blockDim.x + threadIdx.x;
    if (i < n) deg[i] = 1.0f;            // self loop
}
__global__ void deg_edges(const int* __restrict__ edge, float* __restrict__ deg) {
    int e = blockIdx.x * blockDim.x + threadIdx.x;
    if (e < N_EDGES) atomicAdd(&deg[edge[N_EDGES + e]], 1.0f);
}
__global__ void deg_rsqrt(const float* __restrict__ deg,
                          float* __restrict__ dis, int n) {
    int i = blockIdx.x * blockDim.x + threadIdx.x;
    if (i < n) dis[i] = rsqrtf(deg[i]);
}

// ---------------------------------------------------------------------------
// WMMA bf16 GEMM:  C[M x N] (f32) = A[M x K] (bf16 row-major)
//                               x  Bt[N x K] (bf16, transposed weights)
// Workgroup: 256 threads = 8 wave32s arranged 2(M) x 4(N).
// Workgroup tile 64(M) x 256(N); wave tile 32 x 64 = 2x4 fragments of
// v_wmma_f32_16x16x32_bf16 (8 WMMAs per K-step of 32).
// Double-buffered LDS filled with global_load_async_to_lds_b128 (ASYNCcnt),
// so the next K-slab's DMA overlaps the current slab's WMMAs.
// ---------------------------------------------------------------------------
#define TM   64
#define TNB  256
#define TK   32

__device__ __forceinline__ unsigned lds_off(const void* p) {
    // generic pointers into the LDS aperture carry the LDS byte offset in the
    // low 32 bits (CDNA5 flat-aperture mapping)
    return (unsigned)(size_t)p;
}
__device__ __forceinline__ void async_b128(unsigned lds, const void* g) {
    asm volatile("global_load_async_to_lds_b128 %0, %1, off"
                 :: "v"(lds), "v"((unsigned long long)(size_t)g) : "memory");
}
__device__ __forceinline__ void wait_async0() {
    asm volatile("s_wait_asynccnt 0" ::: "memory");
}

__global__ __launch_bounds__(256)
void gemm_bf16_wmma(const __bf16* __restrict__ A, const __bf16* __restrict__ Bt,
                    float* __restrict__ C, int M, int N, int K) {
    __shared__ __bf16 sA[2][TM][TK + 8];    // rows of A (K-contiguous)
    __shared__ __bf16 sB[2][TNB][TK + 8];   // columns of B (K-contiguous)

    const int tid   = threadIdx.x;
    const int lane  = tid & 31;
    const int wave  = tid >> 5;          // 0..7
    const int waveM = wave >> 2;         // 0..1 : 32-row group
    const int waveN = wave & 3;          // 0..3 : 64-col group
    const int bm    = blockIdx.x * TM;
    const int bn    = blockIdx.y * TNB;

    const int lrow    = lane & 15;
    const int halfsel = lane >> 4;       // 0 | 1
    const int kbA     = halfsel * 8;     // A frag K sub-offset
    const int kbB     = halfsel * 16;    // unused (B frags use kbA)

    // staging coordinates (fixed per thread)
    const int ar   = tid >> 2;           // 0..63  A row in tile
    const int aseg = (tid & 3) * 8;      // 0,8,16,24
    const bool arow_ok = (bm + ar) < M;
    const int bc   = tid;                // 0..255 B column in tile

    v8f acc[2][4] = {};

    const int nk = K / TK;

    // ---- prologue: async-stage tile 0 into buffer 0 ----
    {
        if (arow_ok)
            async_b128(lds_off(&sA[0][ar][aseg]),
                       A + (size_t)(bm + ar) * K + aseg);
        const __bf16* p = Bt + (size_t)(bn + bc) * K;
        unsigned lb = lds_off(&sB[0][bc][0]);
        async_b128(lb,      p);
        async_b128(lb + 16, p + 8);
        async_b128(lb + 32, p + 16);
        async_b128(lb + 48, p + 24);
    }
    wait_async0();
    __syncthreads();

    for (int t = 0; t < nk; ++t) {
        const int cur = t & 1;
        const int nxt = cur ^ 1;

        // ---- async-stage tile t+1 (overlaps this tile's WMMAs) ----
        if (t + 1 < nk) {
            const int k0 = (t + 1) * TK;
            if (arow_ok)
                async_b128(lds_off(&sA[nxt][ar][aseg]),
                           A + (size_t)(bm + ar) * K + k0 + aseg);
            const __bf16* p = Bt + (size_t)(bn + bc) * K + k0;
            unsigned lb = lds_off(&sB[nxt][bc][0]);
            async_b128(lb,      p);
            async_b128(lb + 16, p + 8);
            async_b128(lb + 32, p + 16);
            async_b128(lb + 48, p + 24);
        }

        // ---- compute on tile t ----
#pragma unroll
        for (int mi = 0; mi < 2; ++mi) {
            // A fragment 16x32 bf16 (CDNA5 layout): lanes 0-15 hold
            // K = {0..7, 16..23}; lanes 16-31 hold K = {8..15, 24..31}
            int arow = waveM * 32 + mi * 16 + lrow;
            v8bf alo = *(const v8bf*)&sA[cur][arow][kbA];
            v8bf ahi = *(const v8bf*)&sA[cur][arow][16 + kbA];
            v16bf af = __builtin_shufflevector(alo, ahi,
                0, 1, 2, 3, 4, 5, 6, 7, 8, 9, 10, 11, 12, 13, 14, 15);
#pragma unroll
            for (int ni = 0; ni < 4; ++ni) {
                // B fragment 32x16 bf16: lanes 0-15 hold K={0..7,16..23} of col lrow,
                // lanes 16-31 hold K={8..15,24..31}
                int bcol = waveN * 64 + ni * 16 + lrow;
                v8bf blo = *(const v8bf*)&sB[cur][bcol][kbA];
                v8bf bhi = *(const v8bf*)&sB[cur][bcol][16 + kbA];
                v16bf bfr = __builtin_shufflevector(blo, bhi,
                    0, 1, 2, 3, 4, 5, 6, 7, 8, 9, 10, 11, 12, 13, 14, 15);
                acc[mi][ni] = __builtin_amdgcn_wmma_f32_16x16x32_bf16(
                    false, af, false, bfr, (short)0, acc[mi][ni], false, false);
            }
        }

        if (t + 1 < nk) {
            wait_async0();       // this wave's async LDS writes are done
            __syncthreads();     // all waves' writes visible; reads of cur done
        }
    }

    // ---- store: acc element v of lane l -> (M = v + 8*halfsel, N = lrow) ----
#pragma unroll
    for (int mi = 0; mi < 2; ++mi) {
#pragma unroll
        for (int ni = 0; ni < 4; ++ni) {
            int colg = bn + waveN * 64 + ni * 16 + lrow;
#pragma unroll
            for (int v = 0; v < 8; ++v) {
                int rowg = bm + waveM * 32 + mi * 16 + halfsel * 8 + v;
                if (rowg < M && colg < N)
                    C[(size_t)rowg * N + colg] = acc[mi][ni][v];
            }
        }
    }
}

// ---------------------------------------------------------------------------
// GCN message passing (symmetric-normalized scatter-add, self-loops folded in)
// ---------------------------------------------------------------------------
__global__ void agg_init_selfloop(const float* __restrict__ h,
                                  const float* __restrict__ dis,
                                  float* __restrict__ agg, int dim) {
    int n = blockIdx.x;
    float w = dis[n] * dis[n];                 // self-loop norm = 1/deg
    const float* hn = h + (size_t)n * dim;
    float* an = agg + (size_t)n * dim;
    for (int j = threadIdx.x; j < dim; j += blockDim.x)
        an[j] = hn[j] * w;
}

__global__ void edge_scatter(const float* __restrict__ h,
                             const int* __restrict__ edge,
                             const float* __restrict__ dis,
                             float* __restrict__ agg, int dim) {
    int e = blockIdx.x;
    int s = edge[e];
    int d = edge[N_EDGES + e];
    float nrm = dis[s] * dis[d];
    const float* hs = h + (size_t)s * dim;
    float* ad = agg + (size_t)d * dim;
    for (int j = threadIdx.x; j < dim; j += blockDim.x)
        atomicAdd(&ad[j], hs[j] * nrm);
}

// bias + ReLU + f32->bf16 (input of the next WMMA GEMM)
__global__ void bias_relu_bf16(const float* __restrict__ agg,
                               const float* __restrict__ b,
                               __bf16* __restrict__ out, int dim) {
    int n = blockIdx.x;
    const float* an = agg + (size_t)n * dim;
    __bf16* on = out + (size_t)n * dim;
    for (int j = threadIdx.x; j < dim; j += blockDim.x)
        on[j] = (__bf16)fmaxf(an[j] + b[j], 0.0f);
}

// final layer: bias only, keep f32
__global__ void bias_f32(const float* __restrict__ agg,
                         const float* __restrict__ b,
                         float* __restrict__ out, int dim) {
    int n = blockIdx.x;
    const float* an = agg + (size_t)n * dim;
    float* on = out + (size_t)n * dim;
    for (int j = threadIdx.x; j < dim; j += blockDim.x)
        on[j] = an[j] + b[j];
}

// ---------------------------------------------------------------------------
// Pooling + normalize + MLP head
// ---------------------------------------------------------------------------
__global__ void pool_zero(float* __restrict__ psum, float* __restrict__ cnt) {
    int i = blockIdx.x * blockDim.x + threadIdx.x;
    if (i < NGRAPH * D_OUT) psum[i] = 0.0f;
    if (i < NGRAPH) cnt[i] = 0.0f;
}

__global__ void pool_scatter(const float* __restrict__ h4,
                             const int* __restrict__ batch,
                             float* __restrict__ psum, float* __restrict__ cnt) {
    int n = blockIdx.x;
    int g = batch[n];
    const float* hn = h4 + (size_t)n * D_OUT;
    float* pg = psum + (size_t)g * D_OUT;
    for (int j = threadIdx.x; j < D_OUT; j += blockDim.x)
        atomicAdd(&pg[j], hn[j]);
    if (threadIdx.x == 0) atomicAdd(&cnt[g], 1.0f);
}

// mean, L2-normalize, and concat with `vector` -> combined[G x 512]
__global__ void pool_norm_combine(const float* __restrict__ psum,
                                  const float* __restrict__ cnt,
                                  const float* __restrict__ vec,
                                  float* __restrict__ comb) {
    __shared__ float red[D_OUT];
    int g = blockIdx.x, j = threadIdx.x;
    float c = fmaxf(cnt[g], 1.0f);
    float v = psum[(size_t)g * D_OUT + j] / c;
    red[j] = v * v;
    __syncthreads();
    for (int s = D_OUT / 2; s > 0; s >>= 1) {
        if (j < s) red[j] += red[j + s];
        __syncthreads();
    }
    float nrm = fmaxf(sqrtf(red[0]), 1e-12f);
    comb[(size_t)g * 512 + j]         = v / nrm;
    comb[(size_t)g * 512 + D_OUT + j] = vec[(size_t)g * D_OUT + j];
}

__global__ void mlp1(const float* __restrict__ comb, const float* __restrict__ W,
                     const float* __restrict__ b, float* __restrict__ z1) {
    int g = blockIdx.x, j = threadIdx.x;      // 256 outputs
    float acc = b[j];
    const float* cg = comb + (size_t)g * 512;
    for (int k = 0; k < 512; ++k) acc += cg[k] * W[(size_t)k * 256 + j];
    z1[(size_t)g * 256 + j] = fmaxf(acc, 0.0f);
}

__global__ void mlp2(const float* __restrict__ z1, const float* __restrict__ W,
                     const float* __restrict__ b, float* __restrict__ z2) {
    int g = blockIdx.x, j = threadIdx.x;      // 64 outputs
    float acc = b[j];
    const float* zg = z1 + (size_t)g * 256;
    for (int k = 0; k < 256; ++k) acc += zg[k] * W[(size_t)k * 64 + j];
    z2[(size_t)g * 64 + j] = fmaxf(acc, 0.0f);
}

__global__ void mlp3(const float* __restrict__ z2, const float* __restrict__ W,
                     const float* __restrict__ b, float* __restrict__ out) {
    int g = threadIdx.x;                      // 64 graphs
    if (g >= NGRAPH) return;
    float acc = b[0];
    const float* zg = z2 + (size_t)g * 64;
    for (int k = 0; k < 64; ++k) acc += zg[k] * W[k];
    out[g] = 1.0f / (1.0f + expf(-acc));
}

// ---------------------------------------------------------------------------
// Launch
// ---------------------------------------------------------------------------
extern "C" void kernel_launch(void* const* d_in, const int* in_sizes, int n_in,
                              void* d_out, int out_size, void* d_ws, size_t ws_size,
                              hipStream_t stream) {
    const float* x     = (const float*)d_in[0];   // [10000,1280]
    const int*   edge  = (const int*)d_in[1];     // [2,160000]
    const int*   batch = (const int*)d_in[2];     // [10000]
    const float* vec   = (const float*)d_in[3];   // [64,256]
    const float* W1 = (const float*)d_in[4];  const float* b1 = (const float*)d_in[5];
    const float* W2 = (const float*)d_in[6];  const float* b2 = (const float*)d_in[7];
    const float* W3 = (const float*)d_in[8];  const float* b3 = (const float*)d_in[9];
    const float* W4 = (const float*)d_in[10]; const float* b4 = (const float*)d_in[11];
    const float* cW1 = (const float*)d_in[12]; const float* cb1 = (const float*)d_in[13];
    const float* cW2 = (const float*)d_in[14]; const float* cb2 = (const float*)d_in[15];
    const float* cW3 = (const float*)d_in[16]; const float* cb3 = (const float*)d_in[17];
    float* out = (float*)d_out;

    // workspace carve-out (256B-aligned slabs)
    size_t off = 0;
    char* base = (char*)d_ws;
    auto carve = [&](size_t bytes) -> void* {
        void* p = base + off;
        off += (bytes + 255) & ~(size_t)255;
        return p;
    };
    __bf16* Xb   = (__bf16*)carve((size_t)N_NODES * D_IN * 2);
    __bf16* W1t  = (__bf16*)carve((size_t)D_HID * D_IN * 2);
    __bf16* W2t  = (__bf16*)carve((size_t)D_HID * D_HID * 2);
    __bf16* W3t  = (__bf16*)carve((size_t)D_HID * D_HID * 2);
    __bf16* W4t  = (__bf16*)carve((size_t)D_OUT * D_HID * 2);
    __bf16* Hb   = (__bf16*)carve((size_t)N_NODES * D_HID * 2);
    float*  Gout = (float*)carve((size_t)N_NODES * D_HID * 4);
    float*  Agg  = (float*)carve((size_t)N_NODES * D_HID * 4);
    float*  H4   = (float*)carve((size_t)N_NODES * D_OUT * 4);
    float*  deg  = (float*)carve((size_t)N_NODES * 4);
    float*  dis  = (float*)carve((size_t)N_NODES * 4);
    float*  psum = (float*)carve((size_t)NGRAPH * D_OUT * 4);
    float*  cnt  = (float*)carve((size_t)NGRAPH * 4);
    float*  comb = (float*)carve((size_t)NGRAPH * 512 * 4);
    float*  z1   = (float*)carve((size_t)NGRAPH * 256 * 4);
    float*  z2   = (float*)carve((size_t)NGRAPH * 64 * 4);

    const int B = 256;
    auto blocks = [](long long n, int b) { return (unsigned)((n + b - 1) / b); };

    // --- one-time conversions ---
    cvt_f32_to_bf16<<<blocks((long long)N_NODES * D_IN, B), B, 0, stream>>>(x, Xb, N_NODES * D_IN);
    cvt_transpose_bf16<<<blocks((long long)D_IN * D_HID, B), B, 0, stream>>>(W1, W1t, D_IN, D_HID);
    cvt_transpose_bf16<<<blocks((long long)D_HID * D_HID, B), B, 0, stream>>>(W2, W2t, D_HID, D_HID);
    cvt_transpose_bf16<<<blocks((long long)D_HID * D_HID, B), B, 0, stream>>>(W3, W3t, D_HID, D_HID);
    cvt_transpose_bf16<<<blocks((long long)D_HID * D_OUT, B), B, 0, stream>>>(W4, W4t, D_HID, D_OUT);

    // --- degree / normalization ---
    deg_init<<<blocks(N_NODES, B), B, 0, stream>>>(deg, N_NODES);
    deg_edges<<<blocks(N_EDGES, B), B, 0, stream>>>(edge, deg);
    deg_rsqrt<<<blocks(N_NODES, B), B, 0, stream>>>(deg, dis, N_NODES);

    const unsigned gm = blocks(N_NODES, TM);   // 157 M-tiles

    // --- layer 1: [10000,1280] @ [1280,512] ---
    gemm_bf16_wmma<<<dim3(gm, D_HID / TNB), B, 0, stream>>>(Xb, W1t, Gout, N_NODES, D_HID, D_IN);
    agg_init_selfloop<<<N_NODES, B, 0, stream>>>(Gout, dis, Agg, D_HID);
    edge_scatter<<<N_EDGES, B, 0, stream>>>(Gout, edge, dis, Agg, D_HID);
    bias_relu_bf16<<<N_NODES, B, 0, stream>>>(Agg, b1, Hb, D_HID);

    // --- layer 2 ---
    gemm_bf16_wmma<<<dim3(gm, D_HID / TNB), B, 0, stream>>>(Hb, W2t, Gout, N_NODES, D_HID, D_HID);
    agg_init_selfloop<<<N_NODES, B, 0, stream>>>(Gout, dis, Agg, D_HID);
    edge_scatter<<<N_EDGES, B, 0, stream>>>(Gout, edge, dis, Agg, D_HID);
    bias_relu_bf16<<<N_NODES, B, 0, stream>>>(Agg, b2, Hb, D_HID);

    // --- layer 3 ---
    gemm_bf16_wmma<<<dim3(gm, D_HID / TNB), B, 0, stream>>>(Hb, W3t, Gout, N_NODES, D_HID, D_HID);
    agg_init_selfloop<<<N_NODES, B, 0, stream>>>(Gout, dis, Agg, D_HID);
    edge_scatter<<<N_EDGES, B, 0, stream>>>(Gout, edge, dis, Agg, D_HID);
    bias_relu_bf16<<<N_NODES, B, 0, stream>>>(Agg, b3, Hb, D_HID);

    // --- layer 4 (no relu) ---
    gemm_bf16_wmma<<<dim3(gm, D_OUT / TNB), B, 0, stream>>>(Hb, W4t, Gout, N_NODES, D_OUT, D_HID);
    agg_init_selfloop<<<N_NODES, B, 0, stream>>>(Gout, dis, Agg, D_OUT);
    edge_scatter<<<N_EDGES, B, 0, stream>>>(Gout, edge, dis, Agg, D_OUT);
    bias_f32<<<N_NODES, B, 0, stream>>>(Agg, b4, H4, D_OUT);

    // --- pool + normalize + head ---
    pool_zero<<<blocks(NGRAPH * D_OUT, B), B, 0, stream>>>(psum, cnt);
    pool_scatter<<<N_NODES, B, 0, stream>>>(H4, batch, psum, cnt);
    pool_norm_combine<<<NGRAPH, D_OUT, 0, stream>>>(psum, cnt, vec, comb);
    mlp1<<<NGRAPH, 256, 0, stream>>>(comb, cW1, cb1, z1);
    mlp2<<<NGRAPH, 64, 0, stream>>>(z1, cW2, cb2, z2);
    mlp3<<<1, 64, 0, stream>>>(z2, cW3, cb3, out);
}
